// GatedDeltaRecurrence_62242666053984
// MI455X (gfx1250) — compile-verified, matrix operand order checked by reading, not run on hardware
//
#include <hip/hip_runtime.h>
#include <math.h>

// ---------------- problem constants ----------------
#define B_    2
#define T_    2048
#define DM_   1024
#define DKV_  512
#define H_    8
#define KH_   96
#define VH_   192
#define KQT_  768     // H_*KH_
#define VT_   1536    // H_*VH_
#define M_    (B_*T_) // 4096
#define EPS_  1e-6f

typedef __attribute__((ext_vector_type(16))) __bf16 v16bf;
typedef __attribute__((ext_vector_type(8)))  float v8f;
typedef __attribute__((ext_vector_type(4)))  unsigned int u32x4;
typedef __attribute__((ext_vector_type(8)))  int i32x8;
typedef __attribute__((ext_vector_type(4)))  int i32x4;

__device__ __forceinline__ unsigned short f2bf(float f) {
  unsigned int u = __float_as_uint(f);
  unsigned int r = (u + 0x7FFFu + ((u >> 16) & 1u)) >> 16; // RNE
  return (unsigned short)r;
}
__device__ __forceinline__ float silu_f(float x) {
  return x / (1.f + __expf(-x));
}

// ============================================================
// Elementwise fp32 -> bf16 convert.
// ============================================================
__global__ __launch_bounds__(256)
void gdn_cvt_bf16(const float* __restrict__ x, unsigned short* __restrict__ y, int n)
{
  int i = blockIdx.x * 256 + threadIdx.x;
  if (i < n) y[i] = f2bf(x[i]);
}

// ============================================================
// Weight transpose + convert: Wt[n][k] = bf16(W[k][n]).
// LDS-tiled 32x32 to keep both sides coalesced. K,N multiples of 32.
// ============================================================
__global__ __launch_bounds__(256)
void gdn_w_transpose_bf16(const float* __restrict__ W, unsigned short* __restrict__ Wt,
                          int Kdim, int Ndim)
{
  __shared__ float tile[32][33];
  int kb = blockIdx.x * 32, nb = blockIdx.y * 32;
  int tx = threadIdx.x & 31, ty = threadIdx.x >> 5; // 8 rows per pass
#pragma unroll
  for (int i = 0; i < 32; i += 8)
    tile[ty + i][tx] = W[(size_t)(kb + ty + i) * Ndim + (nb + tx)];
  __syncthreads();
#pragma unroll
  for (int i = 0; i < 32; i += 8)
    Wt[(size_t)(nb + ty + i) * Kdim + (kb + tx)] = f2bf(tile[tx][ty + i]);
}

// ============================================================
// GEMM: C[M,N] = act( A[M,K] * W[K,N] ), A bf16 row-major [M][K],
// Wt bf16 pre-transposed [N][K]. Tiles staged via the Tensor Data
// Mover (tensor_load_to_lds, TENSORcnt), consumed as WMMA fragments.
// Block 256 thr = 8 waves; tile 128x64; wave = 32x32 (4 WMMAs/step).
// act: 0 = none, 1 = SiLU.
// ============================================================
__global__ __launch_bounds__(256)
void gdn_gemm_wmma(const unsigned short* __restrict__ A,
                   const unsigned short* __restrict__ Wt,
                   float* __restrict__ C, int Mdim, int Ndim, int Kdim, int act)
{
  __shared__ unsigned short lA[128 * 32]; // [m][k] bf16, 8 KB
  __shared__ unsigned short lB[64 * 32];  // [n][k] bf16, 4 KB

  const int tid  = threadIdx.x;
  const int wave = tid >> 5;
  const int lane = tid & 31;
  const int wr   = wave >> 1;   // 0..3 : 32-row slab
  const int wc   = wave & 1;    // 0..1 : 32-col slab
  const int bm   = blockIdx.x * 128;
  const int bn   = blockIdx.y * 64;
  const int i16  = lane & 15;
  const int hi   = lane >> 4;

  // LDS byte offsets (generic shared address truncates to LDS offset)
  const unsigned ldsA = (unsigned)(size_t)&lA[0];
  const unsigned ldsB = (unsigned)(size_t)&lB[0];

  // ---- loop-invariant TDM D# group1 (256b) per ISA 08 §8.4 ----
  // w0: workgroup_mask=0, data_size=1 (2 bytes)
  // tensor_dim0=K, tensor_dim1=rows, tile_dim0=32, tile_dim1=rows/tile,
  // tensor_dim0_stride=K. Groups 2/3 zero (2-D tensor).
  i32x8 g1A, g1B;
  g1A[0] = 0x00010000;
  g1A[1] = (int)(((unsigned)Kdim & 0xFFFFu) << 16);
  g1A[2] = (int)(((unsigned)Kdim >> 16) | (((unsigned)Mdim & 0xFFFFu) << 16));
  g1A[3] = (int)(((unsigned)Mdim >> 16) | (32u << 16));   // tile_dim0 = 32
  g1A[4] = 128;                                           // tile_dim1 = 128
  g1A[5] = Kdim;                                          // stride0 lo32
  g1A[6] = 0;
  g1A[7] = 0;
  g1B = g1A;
  g1B[2] = (int)(((unsigned)Kdim >> 16) | (((unsigned)Ndim & 0xFFFFu) << 16));
  g1B[3] = (int)(((unsigned)Ndim >> 16) | (32u << 16));
  g1B[4] = 64;                                            // tile_dim1 = 64
  const i32x4 z4 = {0, 0, 0, 0};
  const i32x8 z8 = {0, 0, 0, 0, 0, 0, 0, 0};

  v8f acc00 = {}, acc01 = {}, acc10 = {}, acc11 = {};

  for (int k0 = 0; k0 < Kdim; k0 += 32) {
    if (tid < 32) { // wave 0 drives the TDM
      unsigned long long gA = (unsigned long long)(size_t)A +
                              (((size_t)bm * (size_t)Kdim + (size_t)k0) << 1);
      unsigned long long gB = (unsigned long long)(size_t)Wt +
                              (((size_t)bn * (size_t)Kdim + (size_t)k0) << 1);
      // group0: count=1 | lds_addr | global_addr[56:0] | type=2 (bits 127:126)
      u32x4 g0A = { 1u, ldsA, (unsigned)gA,
                    ((unsigned)(gA >> 32) & 0x01FFFFFFu) | 0x80000000u };
      u32x4 g0B = { 1u, ldsB, (unsigned)gB,
                    ((unsigned)(gB >> 32) & 0x01FFFFFFu) | 0x80000000u };
      __builtin_amdgcn_tensor_load_to_lds(g0A, g1A, z4, z4, z8, 0);
      __builtin_amdgcn_tensor_load_to_lds(g0B, g1B, z4, z4, z8, 0);
      __builtin_amdgcn_s_wait_tensorcnt(0);
    }
    __syncthreads();

    // fragments per CDNA5 16-bit layout: lane i16 = M/N index,
    // dwords at k/2 in {hi*4+j} and {8+hi*4+j} -> two b128 LDS reads each
    union { v16bf v; unsigned int d[8]; } fa0, fa1, fb0, fb1;
    const unsigned int* rA0 = (const unsigned int*)&lA[(wr * 32 + i16) * 32];
    const unsigned int* rA1 = (const unsigned int*)&lA[(wr * 32 + 16 + i16) * 32];
    const unsigned int* rB0 = (const unsigned int*)&lB[(wc * 32 + i16) * 32];
    const unsigned int* rB1 = (const unsigned int*)&lB[(wc * 32 + 16 + i16) * 32];
#pragma unroll
    for (int j = 0; j < 4; ++j) {
      fa0.d[j] = rA0[hi * 4 + j];  fa0.d[j + 4] = rA0[8 + hi * 4 + j];
      fa1.d[j] = rA1[hi * 4 + j];  fa1.d[j + 4] = rA1[8 + hi * 4 + j];
      fb0.d[j] = rB0[hi * 4 + j];  fb0.d[j + 4] = rB0[8 + hi * 4 + j];
      fb1.d[j] = rB1[hi * 4 + j];  fb1.d[j + 4] = rB1[8 + hi * 4 + j];
    }
    acc00 = __builtin_amdgcn_wmma_f32_16x16x32_bf16(false, fa0.v, false, fb0.v,
                                                    (short)0, acc00, false, false);
    acc01 = __builtin_amdgcn_wmma_f32_16x16x32_bf16(false, fa0.v, false, fb1.v,
                                                    (short)0, acc01, false, false);
    acc10 = __builtin_amdgcn_wmma_f32_16x16x32_bf16(false, fa1.v, false, fb0.v,
                                                    (short)0, acc10, false, false);
    acc11 = __builtin_amdgcn_wmma_f32_16x16x32_bf16(false, fa1.v, false, fb1.v,
                                                    (short)0, acc11, false, false);
    __syncthreads();
  }

  // C/D layout: VGPR r holds M=r (lanes 0-15) / M=8+r (lanes 16-31), N=lane%16
#pragma unroll
  for (int r = 0; r < 8; ++r) {
    int row0 = bm + wr * 32 + r + hi * 8;
    int row1 = row0 + 16;
    int col  = bn + wc * 32 + i16;
    float v00 = acc00[r], v01 = acc01[r], v10 = acc10[r], v11 = acc11[r];
    if (act == 1) { v00 = silu_f(v00); v01 = silu_f(v01);
                    v10 = silu_f(v10); v11 = silu_f(v11); }
    C[(size_t)row0 * Ndim + col]      = v00;
    C[(size_t)row0 * Ndim + col + 16] = v01;
    C[(size_t)row1 * Ndim + col]      = v10;
    C[(size_t)row1 * Ndim + col + 16] = v11;
  }
}

// ============================================================
// a/b projections (N=8 each) + g = -softplus(dt_bias+a)*exp(A_log),
// beta = sigmoid(b). One thread per (m,h). Output layout (B,H,T).
// ============================================================
__global__ __launch_bounds__(256)
void gdn_proj_gb(const float* __restrict__ x,
                 const float* __restrict__ a_w, const float* __restrict__ a_b,
                 const float* __restrict__ b_w, const float* __restrict__ b_b,
                 const float* __restrict__ A_log, const float* __restrict__ dt_bias,
                 float* __restrict__ g, float* __restrict__ beta)
{
  int idx = blockIdx.x * blockDim.x + threadIdx.x; // over M_*H_
  if (idx >= M_ * H_) return;
  int h = idx % H_;
  int m = idx / H_;
  const float* xr = x + (size_t)m * DM_;
  float sa = 0.f, sb = 0.f;
  for (int k = 0; k < DM_; ++k) {
    float xv = xr[k];
    sa += xv * a_w[(size_t)k * H_ + h];
    sb += xv * b_w[(size_t)k * H_ + h];
  }
  sa += a_b[h];
  sb += b_b[h];
  float z  = dt_bias[h] + sa;
  float sp = (z > 20.f) ? z : log1pf(__expf(z));
  float gg = -sp * __expf(A_log[h]);
  float bt = 1.f / (1.f + __expf(-sb));
  int bb = m / T_, t = m % T_;
  size_t o = ((size_t)(bb * H_ + h)) * T_ + t;
  g[o]    = gg;
  beta[o] = bt;
}

// ============================================================
// Depthwise causal conv (K=4, left pad 3) + SiLU. Thread per (b,t,c).
// ============================================================
__global__ __launch_bounds__(256)
void gdn_conv_silu(const float* __restrict__ xin, const float* __restrict__ w,
                   const float* __restrict__ bias, float* __restrict__ y, int C)
{
  size_t idx = (size_t)blockIdx.x * blockDim.x + threadIdx.x; // over B*T*C
  if (idx >= (size_t)M_ * C) return;
  int    c  = (int)(idx % C);
  size_t bt = idx / C;
  int    t  = (int)(bt % T_);
  const float* wr = w + (size_t)c * 4;
  float acc = bias[c];
#pragma unroll
  for (int i = 0; i < 4; ++i) {
    int ts = t - 3 + i;
    if (ts >= 0) acc += wr[i] * xin[(bt - 3 + i) * C + c];
  }
  y[idx] = silu_f(acc);
}

// ============================================================
// Head split + L2-norm for q/k (and head split for v).
// grid.x over (b*H+h)*T+t rows, grid.y: 0=q, 1=k, 2=v.
// ============================================================
__global__ __launch_bounds__(192)
void gdn_prep_heads(const float* __restrict__ qc, const float* __restrict__ kc,
                    const float* __restrict__ vc, float* __restrict__ qn,
                    float* __restrict__ kn, float* __restrict__ vh)
{
  int row   = blockIdx.x;        // (b*H + h)*T + t
  int which = blockIdx.y;
  int t  = row % T_;
  int bh = row / T_;
  int h  = bh % H_;
  int b  = bh / H_;
  int c  = threadIdx.x;

  if (which == 2) {
    for (int i = c; i < VH_; i += 192)
      vh[(size_t)row * VH_ + i] = vc[((size_t)(b * T_ + t)) * VT_ + h * VH_ + i];
    return;
  }
  const float* src = (which == 0) ? qc : kc;
  float*       dst = (which == 0) ? qn : kn;
  __shared__ float red[128];
  float x = 0.f;
  if (c < KH_) x = src[((size_t)(b * T_ + t)) * KQT_ + h * KH_ + c];
  if (c < 128) red[c] = (c < KH_) ? x * x : 0.f;
  __syncthreads();
#pragma unroll
  for (int s = 64; s > 0; s >>= 1) {
    if (c < s) red[c] += red[c + s];
    __syncthreads();
  }
  float scale = 1.f / (sqrtf(red[0]) + EPS_);
  if (which == 0) scale *= 0.10206207261596575f; // KH^-0.5 = 1/sqrt(96)
  if (c < KH_) dst[(size_t)row * KH_ + c] = x * scale;
}

// ============================================================
// Delta-rule scan. State columns S[:,v] are independent:
// 16 blocks (b,h) x 192 threads (one v column each), S[96] in VGPRs.
// k_t/q_t broadcast via LDS each step.
// ============================================================
__global__ __launch_bounds__(192)
void gdn_delta_scan(const float* __restrict__ qn, const float* __restrict__ kn,
                    const float* __restrict__ vh, const float* __restrict__ g,
                    const float* __restrict__ beta, float* __restrict__ o)
{
  const int bh = blockIdx.x;   // 0..15
  const int c  = threadIdx.x;  // v column 0..191
  float S[KH_];
#pragma unroll
  for (int i = 0; i < KH_; ++i) S[i] = 0.f;

  __shared__ float sk[KH_];
  __shared__ float sq[KH_];
  const size_t base = (size_t)bh * T_;

  for (int t = 0; t < T_; ++t) {
    __syncthreads();
    if (c < KH_)       sk[c]       = kn[(base + t) * KH_ + c];
    else               sq[c - KH_] = qn[(base + t) * KH_ + (c - KH_)];
    __syncthreads();

    const float eg = __expf(g[base + t]);
    const float bt = beta[base + t];
    const float vt = vh[(base + t) * VH_ + c];

    float dot = 0.f;
#pragma unroll
    for (int i = 0; i < KH_; ++i) {
      float s = S[i] * eg;
      S[i] = s;
      dot += s * sk[i];
    }
    const float dv = bt * (vt - dot);
    float out = 0.f;
#pragma unroll
    for (int i = 0; i < KH_; ++i) {
      float s = S[i] + sk[i] * dv;
      S[i] = s;
      out += s * sq[i];
    }
    o[(base + t) * VH_ + c] = out;
  }
}

// ============================================================
// RMS-norm over VT, * post_norm_w, * gate. Head-major -> (B,T,VT).
// One 256-thread block per (b,t) row, 6 elems/thread.
// ============================================================
__global__ __launch_bounds__(256)
void gdn_mix(const float* __restrict__ o_scan, const float* __restrict__ gate,
             const float* __restrict__ pnw, float* __restrict__ o_mix)
{
  int m = blockIdx.x;  // b*T + t
  int b = m / T_, t = m % T_;
  __shared__ float red[256];
  float loc[6];
  float ss = 0.f;
#pragma unroll
  for (int i = 0; i < 6; ++i) {
    int n  = threadIdx.x + 256 * i;   // 0..1535
    int h  = n / VH_, v = n % VH_;
    float x = o_scan[(((size_t)(b * H_ + h)) * T_ + t) * VH_ + v];
    loc[i] = x;
    ss += x * x;
  }
  red[threadIdx.x] = ss;
  __syncthreads();
#pragma unroll
  for (int s = 128; s > 0; s >>= 1) {
    if (threadIdx.x < s) red[threadIdx.x] += red[threadIdx.x + s];
    __syncthreads();
  }
  float scale = rsqrtf(red[0] * (1.f / VT_) + EPS_);
#pragma unroll
  for (int i = 0; i < 6; ++i) {
    int n = threadIdx.x + 256 * i;
    o_mix[(size_t)m * VT_ + n] = loc[i] * scale * pnw[n] * gate[(size_t)m * VT_ + n];
  }
}

// ============================================================
// launcher
// ============================================================
extern "C" void kernel_launch(void* const* d_in, const int* in_sizes, int n_in,
                              void* d_out, int out_size, void* d_ws, size_t ws_size,
                              hipStream_t stream)
{
  const float* x        = (const float*)d_in[0];
  const float* c_kv     = (const float*)d_in[1];
  const float* w_q      = (const float*)d_in[2];
  const float* w_k      = (const float*)d_in[3];
  const float* w_v      = (const float*)d_in[4];
  const float* conv_q_w = (const float*)d_in[5];
  const float* conv_q_b = (const float*)d_in[6];
  const float* conv_k_w = (const float*)d_in[7];
  const float* conv_k_b = (const float*)d_in[8];
  const float* conv_v_w = (const float*)d_in[9];
  const float* conv_v_b = (const float*)d_in[10];
  const float* a_proj_w = (const float*)d_in[11];
  const float* a_proj_b = (const float*)d_in[12];
  const float* A_log    = (const float*)d_in[13];
  const float* dt_bias  = (const float*)d_in[14];
  const float* b_proj_w = (const float*)d_in[15];
  const float* b_proj_b = (const float*)d_in[16];
  const float* g_proj_w = (const float*)d_in[17];
  const float* pnw      = (const float*)d_in[18];
  const float* w_o      = (const float*)d_in[19];
  float* out = (float*)d_out;

  // ---------------- workspace layout ----------------
  const size_t SZ_QK = (size_t)M_ * KQT_; // 3,145,728 floats
  const size_t SZ_V  = (size_t)M_ * VT_;  // 6,291,456 floats
  float* ws = (float*)d_ws;
  float* q_pre = ws;                          // later reused as qn
  float* k_pre = q_pre + SZ_QK;               // later reused as kn
  float* v_pre = k_pre + SZ_QK;               // later reused as vh (heads)
  float* qc    = v_pre + SZ_V;                // qc+kc later reused as o_scan
  float* kc    = qc + SZ_QK;
  float* vc    = kc + SZ_QK;                  // later reused as o_mix
  float* gbuf  = vc + SZ_V;                   // (B,H,T)
  float* bbuf  = gbuf + (size_t)B_ * H_ * T_;
  float* gate  = bbuf + (size_t)B_ * H_ * T_;
  float* fend  = gate + SZ_V;
  float* qn = q_pre; float* kn = k_pre; float* vhd = v_pre;
  float* o_scan = qc;      // SZ_V floats == 2*SZ_QK (spans qc+kc exactly)
  float* o_mix  = vc;

  // bf16 region
  unsigned short* xb    = (unsigned short*)fend;
  unsigned short* ckvb  = xb    + (size_t)M_ * DM_;
  unsigned short* omixb = ckvb  + (size_t)M_ * DKV_;
  unsigned short* wqT   = omixb + (size_t)M_ * VT_;   // [KQT_][DM_]
  unsigned short* wkT   = wqT   + (size_t)KQT_ * DM_; // [KQT_][DKV_]
  unsigned short* wvT   = wkT   + (size_t)KQT_ * DKV_;// [VT_][DKV_]
  unsigned short* gpT   = wvT   + (size_t)VT_ * DKV_; // [VT_][DM_]
  unsigned short* woT   = gpT   + (size_t)VT_ * DM_;  // [DM_][VT_]

  // 0) one-time bf16 conversion / weight transposition
  gdn_cvt_bf16<<<(M_ * DM_ + 255) / 256, 256, 0, stream>>>(x, xb, M_ * DM_);
  gdn_cvt_bf16<<<(M_ * DKV_ + 255) / 256, 256, 0, stream>>>(c_kv, ckvb, M_ * DKV_);
  gdn_w_transpose_bf16<<<dim3(DM_ / 32, KQT_ / 32), 256, 0, stream>>>(w_q, wqT, DM_, KQT_);
  gdn_w_transpose_bf16<<<dim3(DKV_ / 32, KQT_ / 32), 256, 0, stream>>>(w_k, wkT, DKV_, KQT_);
  gdn_w_transpose_bf16<<<dim3(DKV_ / 32, VT_ / 32), 256, 0, stream>>>(w_v, wvT, DKV_, VT_);
  gdn_w_transpose_bf16<<<dim3(DM_ / 32, VT_ / 32), 256, 0, stream>>>(g_proj_w, gpT, DM_, VT_);
  gdn_w_transpose_bf16<<<dim3(VT_ / 32, DM_ / 32), 256, 0, stream>>>(w_o, woT, VT_, DM_);

  // 1) projection GEMMs (TDM-staged bf16 WMMA, fp32 accumulate)
  gdn_gemm_wmma<<<dim3(M_ / 128, KQT_ / 64), 256, 0, stream>>>(xb,   wqT, q_pre, M_, KQT_, DM_,  0);
  gdn_gemm_wmma<<<dim3(M_ / 128, KQT_ / 64), 256, 0, stream>>>(ckvb, wkT, k_pre, M_, KQT_, DKV_, 0);
  gdn_gemm_wmma<<<dim3(M_ / 128, VT_  / 64), 256, 0, stream>>>(ckvb, wvT, v_pre, M_, VT_,  DKV_, 0);
  gdn_gemm_wmma<<<dim3(M_ / 128, VT_  / 64), 256, 0, stream>>>(xb,   gpT, gate,  M_, VT_,  DM_,  1); // SiLU

  // 2) a/b projections -> g, beta
  gdn_proj_gb<<<(M_ * H_ + 255) / 256, 256, 0, stream>>>(x, a_proj_w, a_proj_b,
      b_proj_w, b_proj_b, A_log, dt_bias, gbuf, bbuf);

  // 3) depthwise causal conv + SiLU
  gdn_conv_silu<<<(unsigned)(((size_t)M_ * KQT_ + 255) / 256), 256, 0, stream>>>(q_pre, conv_q_w, conv_q_b, qc, KQT_);
  gdn_conv_silu<<<(unsigned)(((size_t)M_ * KQT_ + 255) / 256), 256, 0, stream>>>(k_pre, conv_k_w, conv_k_b, kc, KQT_);
  gdn_conv_silu<<<(unsigned)(((size_t)M_ * VT_  + 255) / 256), 256, 0, stream>>>(v_pre, conv_v_w, conv_v_b, vc, VT_);

  // 4) head split + L2 norm (overwrites q_pre/k_pre/v_pre with qn/kn/vhd)
  gdn_prep_heads<<<dim3(B_ * H_ * T_, 3), 192, 0, stream>>>(qc, kc, vc, qn, kn, vhd);

  // 5) delta-rule recurrent scan (register-resident state columns)
  gdn_delta_scan<<<B_ * H_, 192, 0, stream>>>(qn, kn, vhd, gbuf, bbuf, o_scan);

  // 6) RMS-norm + post_norm_w + gate
  gdn_mix<<<M_, 256, 0, stream>>>(o_scan, gate, pnw, o_mix);

  // 7) final output GEMM -> d_out
  gdn_cvt_bf16<<<(M_ * VT_ + 255) / 256, 256, 0, stream>>>(o_mix, omixb, M_ * VT_);
  gdn_gemm_wmma<<<dim3(M_ / 128, DM_ / 64), 256, 0, stream>>>(omixb, woT, out, M_, DM_, VT_, 0);

  (void)in_sizes; (void)n_in; (void)out_size; (void)ws_size;
}
